// GraphFusionEncoder_48490180772598
// MI455X (gfx1250) — matrix operand order repaired and checked by reading, not compiled, not run on hardware
//
#include <hip/hip_runtime.h>
#include <math.h>

// Problem constants (match reference)
#define B_  32
#define NN  512
#define H_  768
#define R_  5
#define L_  2
#define FF_ 1536

constexpr int BLK_M = 128, BLK_N = 64, BLK_K = 32, NT = 128;
constexpr int LDA_S = 40;  // LDS row stride (elems) for A tile [128][32]; 80B = 16B-aligned rows
constexpr int LDB_S = 42;  // LDS row stride (elems) for Bt tile [64][32]

typedef __attribute__((ext_vector_type(16))) _Float16 v16h;
typedef __attribute__((ext_vector_type(8)))  float    v8f;

union FragU { v16h h; unsigned int u[8]; };

__device__ __forceinline__ v8f vzero8() {
  v8f z;
#pragma unroll
  for (int i = 0; i < 8; ++i) z[i] = 0.f;
  return z;
}

// ---------- global -> LDS tile loaders ----------
// A tile 128x32 f16 via gfx1250 async DMA (tracked on ASYNCcnt, no VGPR round-trip).
__device__ __forceinline__ void ldA_h_async(const _Float16* __restrict__ A, int lda,
                                            int row0, int k0, _Float16* As, int t) {
  // 128x32 f16 = 512 chunks of 16B; 4 per thread, consecutive lanes -> consecutive chunks
#pragma unroll
  for (int i = 0; i < 4; ++i) {
    int c4 = i * NT + t;
    int r  = c4 >> 2;
    int co = (c4 & 3) * 8;  // elem offset within row
    unsigned long gaddr = (unsigned long)(const void*)(A + (size_t)(row0 + r) * lda + k0 + co);
    unsigned int  laddr = (unsigned int)(size_t)(const void*)(As + r * LDA_S + co);
    asm volatile("global_load_async_to_lds_b128 %0, %1, off"
                 :: "v"(laddr), "v"(gaddr) : "memory");
  }
}

// A tile 128x32 from f32 source (adjacency), converting to f16 while staging.
__device__ __forceinline__ void ldA_f32(const float* __restrict__ A, int lda,
                                        int row0, int k0, _Float16* As, int t) {
#pragma unroll
  for (int i = 0; i < 8; ++i) {
    int c4 = i * NT + t;            // float4 chunk index (1024 total)
    int r  = c4 >> 3;
    int co = (c4 & 7) * 4;
    const float4 f = *(const float4*)(A + (size_t)(row0 + r) * lda + k0 + co);
    union { _Float16 h[4]; uint2 u; } pk;
    pk.h[0] = (_Float16)f.x; pk.h[1] = (_Float16)f.y;
    pk.h[2] = (_Float16)f.z; pk.h[3] = (_Float16)f.w;
    *(uint2*)(As + r * LDA_S + co) = pk.u;
  }
}

// B tile: 32x64 f16 row-major source (ldb elems), stored transposed Bt[n][k].
__device__ __forceinline__ void ldB_h(const _Float16* __restrict__ Bm, int ldb,
                                      int k0, int col0, _Float16* Bs, int t) {
#pragma unroll
  for (int i = 0; i < 8; ++i) {
    int flat = i * NT + t;          // dword index over 32x64 elems
    int k = flat >> 5;
    int n = (flat & 31) * 2;
    unsigned int v = *(const unsigned int*)(Bm + (size_t)(k0 + k) * ldb + col0 + n);
    union { unsigned int u; _Float16 h[2]; } cv; cv.u = v;
    Bs[(n + 0) * LDB_S + k] = cv.h[0];
    Bs[(n + 1) * LDB_S + k] = cv.h[1];
  }
}

// ---------- LDS -> WMMA fragments (contiguous k = 16*(lane/16) + e, same order for A and B) ----------
__device__ __forceinline__ v16h load_fragA(const _Float16* As, int lane) {
  int m = lane & 15, g = lane >> 4;
  const _Float16* rp = As + m * LDA_S;
  FragU f;
#pragma unroll
  for (int v = 0; v < 8; ++v) {
    int k0 = 2 * v + 16 * g;   // K(e)=e+16g
    f.u[v] = *(const unsigned int*)(rp + k0);
  }
  return f.h;
}

__device__ __forceinline__ v16h load_fragB(const _Float16* Bs, int nTile, int lane) {
  int n = nTile + (lane & 15), g = lane >> 4;
  const _Float16* cp = Bs + n * LDB_S + 16 * g;   // K(e)=e+16g
  FragU f;
#pragma unroll
  for (int v = 0; v < 8; ++v) f.u[v] = *(const unsigned int*)(cp + 2 * v);
  return f.h;
}

#define WMMA_F16(a, b, c) \
  __builtin_amdgcn_wmma_f32_16x16x32_f16(false, (a), false, (b), (short)0, (c), false, false)

// Wave computes a 32x64 strip: 2 A-frags x 4 B-frags = 8 WMMA per K-step.
__device__ __forceinline__ void wmma_step2(const _Float16* As, const _Float16* Bs,
                                           int wave, int lane, v8f acc[2][4]) {
  v16h a0 = load_fragA(As + (wave * 32 +  0) * LDA_S, lane);
  v16h a1 = load_fragA(As + (wave * 32 + 16) * LDA_S, lane);
  v16h b0 = load_fragB(Bs,  0, lane);
  v16h b1 = load_fragB(Bs, 16, lane);
  v16h b2 = load_fragB(Bs, 32, lane);
  v16h b3 = load_fragB(Bs, 48, lane);
  acc[0][0] = WMMA_F16(a0, b0, acc[0][0]);
  acc[0][1] = WMMA_F16(a0, b1, acc[0][1]);
  acc[0][2] = WMMA_F16(a0, b2, acc[0][2]);
  acc[0][3] = WMMA_F16(a0, b3, acc[0][3]);
  acc[1][0] = WMMA_F16(a1, b0, acc[1][0]);
  acc[1][1] = WMMA_F16(a1, b1, acc[1][1]);
  acc[1][2] = WMMA_F16(a1, b2, acc[1][2]);
  acc[1][3] = WMMA_F16(a1, b3, acc[1][3]);
}

#define WAIT_ASYNC() asm volatile("s_wait_asynccnt 0" ::: "memory")

// ---------- G1: agg = (adj @ x_h) * deg_inv  -> f16 ----------
__global__ __launch_bounds__(NT) void gemm_agg_kernel(
    const float* __restrict__ adj, const _Float16* __restrict__ xh,
    const float* __restrict__ deginv, _Float16* __restrict__ aggh) {
  __shared__ __align__(16) _Float16 As[2][BLK_M * LDA_S];
  __shared__ __align__(16) _Float16 Bs[2][BLK_N * LDB_S];
  int t = threadIdx.x, lane = t & 31, wave = t >> 5;
  int br = blockIdx.z, b = br / R_;
  int row0 = blockIdx.y * BLK_M, col0 = blockIdx.x * BLK_N;
  const float*    Ab = adj + (size_t)br * NN * NN;
  const _Float16* Bb = xh  + (size_t)b  * NN * H_;
  v8f acc[2][4];
  for (int mi = 0; mi < 2; ++mi) for (int j = 0; j < 4; ++j) acc[mi][j] = vzero8();
  int buf = 0;
  ldA_f32(Ab, NN, row0, 0, As[0], t);
  ldB_h(Bb, H_, 0, col0, Bs[0], t);
  __syncthreads();
  for (int k0 = 0; k0 < NN; k0 += BLK_K) {
    int nk = k0 + BLK_K;
    if (nk < NN) {
      ldA_f32(Ab, NN, row0, nk, As[buf ^ 1], t);
      ldB_h(Bb, H_, nk, col0, Bs[buf ^ 1], t);
    }
    wmma_step2(As[buf], Bs[buf], wave, lane, acc);
    __syncthreads();
    buf ^= 1;
  }
  int g = lane >> 4, cn = lane & 15;
#pragma unroll
  for (int mi = 0; mi < 2; ++mi)
#pragma unroll
    for (int i = 0; i < 8; ++i) {
      int row = row0 + wave * 32 + mi * 16 + i + 8 * g;
      float s = deginv[(size_t)br * NN + row];
#pragma unroll
      for (int j = 0; j < 4; ++j) {
        int col = col0 + j * 16 + cn;
        aggh[((size_t)br * NN + row) * H_ + col] = (_Float16)(acc[mi][j][i] * s);
      }
    }
}

// ---------- G2: merged = sum_r w[b,r] * relu(agg[b,r] @ relW[r] + rel_b[r]) -> f16 ----------
__global__ __launch_bounds__(NT) void gemm_merge_kernel(
    const _Float16* __restrict__ aggh, const _Float16* __restrict__ relWh,
    const float* __restrict__ relb, const float* __restrict__ w,
    _Float16* __restrict__ mergedh) {
  __shared__ __align__(16) _Float16 As[2][BLK_M * LDA_S];
  __shared__ __align__(16) _Float16 Bs[2][BLK_N * LDB_S];
  int t = threadIdx.x, lane = t & 31, wave = t >> 5;
  int b = blockIdx.z;
  int row0 = blockIdx.y * BLK_M, col0 = blockIdx.x * BLK_N;
  int g = lane >> 4, cn = lane & 15;
  v8f macc[2][4];
  for (int mi = 0; mi < 2; ++mi) for (int j = 0; j < 4; ++j) macc[mi][j] = vzero8();
  for (int r = 0; r < R_; ++r) {
    const _Float16* Ab = aggh  + ((size_t)(b * R_ + r) * NN) * H_;
    const _Float16* Bw = relWh + (size_t)r * H_ * H_;
    v8f acc[2][4];
    for (int mi = 0; mi < 2; ++mi) for (int j = 0; j < 4; ++j) acc[mi][j] = vzero8();
    int buf = 0;
    ldA_h_async(Ab, H_, row0, 0, As[0], t);
    ldB_h(Bw, H_, 0, col0, Bs[0], t);
    WAIT_ASYNC();
    __syncthreads();
    for (int k0 = 0; k0 < H_; k0 += BLK_K) {
      int nk = k0 + BLK_K;
      if (nk < H_) {
        ldA_h_async(Ab, H_, row0, nk, As[buf ^ 1], t);
        ldB_h(Bw, H_, nk, col0, Bs[buf ^ 1], t);
      }
      wmma_step2(As[buf], Bs[buf], wave, lane, acc);
      WAIT_ASYNC();
      __syncthreads();
      buf ^= 1;
    }
    float wr = w[b * R_ + r];
#pragma unroll
    for (int j = 0; j < 4; ++j) {
      int col = col0 + j * 16 + cn;
      float bv = relb[r * H_ + col];
#pragma unroll
      for (int mi = 0; mi < 2; ++mi)
#pragma unroll
        for (int i = 0; i < 8; ++i)
          macc[mi][j][i] += wr * fmaxf(acc[mi][j][i] + bv, 0.f);
    }
    __syncthreads();  // before next r reuses LDS buffers
  }
#pragma unroll
  for (int mi = 0; mi < 2; ++mi)
#pragma unroll
    for (int i = 0; i < 8; ++i) {
      int row = row0 + wave * 32 + mi * 16 + i + 8 * g;
#pragma unroll
      for (int j = 0; j < 4; ++j) {
        int col = col0 + j * 16 + cn;
        mergedh[((size_t)b * NN + row) * H_ + col] = (_Float16)macc[mi][j][i];
      }
    }
}

// ---------- generic GEMM: out = [relu](A@B + bias [+ resid]) -> f32 and/or f16 ----------
__global__ __launch_bounds__(NT) void gemm_bias_kernel(
    const _Float16* __restrict__ A, const _Float16* __restrict__ Bm,
    const float* __restrict__ bias, const float* __restrict__ resid,
    float* __restrict__ outF, _Float16* __restrict__ outH,
    int M, int N, int K,
    long aBatch, long rBatch, long oBatch, int relu) {
  __shared__ __align__(16) _Float16 As[2][BLK_M * LDA_S];
  __shared__ __align__(16) _Float16 Bs[2][BLK_N * LDB_S];
  int t = threadIdx.x, lane = t & 31, wave = t >> 5;
  int b = blockIdx.z;
  int row0 = blockIdx.y * BLK_M, col0 = blockIdx.x * BLK_N;
  const _Float16* Ab = A + (size_t)b * aBatch;
  v8f acc[2][4];
  for (int mi = 0; mi < 2; ++mi) for (int j = 0; j < 4; ++j) acc[mi][j] = vzero8();
  int buf = 0;
  ldA_h_async(Ab, K, row0, 0, As[0], t);
  ldB_h(Bm, N, 0, col0, Bs[0], t);
  WAIT_ASYNC();
  __syncthreads();
  for (int k0 = 0; k0 < K; k0 += BLK_K) {
    int nk = k0 + BLK_K;
    if (nk < K) {
      ldA_h_async(Ab, K, row0, nk, As[buf ^ 1], t);
      ldB_h(Bm, N, nk, col0, Bs[buf ^ 1], t);
    }
    wmma_step2(As[buf], Bs[buf], wave, lane, acc);
    WAIT_ASYNC();
    __syncthreads();
    buf ^= 1;
  }
  int g = lane >> 4, cn = lane & 15;
#pragma unroll
  for (int mi = 0; mi < 2; ++mi)
#pragma unroll
    for (int i = 0; i < 8; ++i) {
      int row = row0 + wave * 32 + mi * 16 + i + 8 * g;
#pragma unroll
      for (int j = 0; j < 4; ++j) {
        int col = col0 + j * 16 + cn;
        float v = acc[mi][j][i] + (bias ? bias[col] : 0.f);
        if (resid) v += resid[(size_t)b * rBatch + (size_t)row * N + col];
        if (relu) v = fmaxf(v, 0.f);
        size_t oi = (size_t)b * oBatch + (size_t)row * N + col;
        if (outF) outF[oi] = v;
        if (outH) outH[oi] = (_Float16)v;
      }
    }
}

// ---------- LayerNorm over H=768, wave-per-row (wave32, 24 elems/lane) ----------
__global__ __launch_bounds__(256) void ln_kernel(
    const float* __restrict__ in, const float* __restrict__ gamma,
    const float* __restrict__ beta, float* __restrict__ outF,
    _Float16* __restrict__ outH, int nrows) {
  int wave = threadIdx.x >> 5, lane = threadIdx.x & 31;
  int row = blockIdx.x * 8 + wave;
  if (row >= nrows) return;
  const float* p = in + (size_t)row * H_;
  float v[24], s = 0.f, s2 = 0.f;
#pragma unroll
  for (int i = 0; i < 24; ++i) { v[i] = p[lane + 32 * i]; s += v[i]; s2 += v[i] * v[i]; }
  for (int off = 16; off > 0; off >>= 1) { s += __shfl_xor(s, off); s2 += __shfl_xor(s2, off); }
  float mu  = s * (1.f / H_);
  float var = s2 * (1.f / H_) - mu * mu;
  float inv = rsqrtf(var + 1e-5f);
#pragma unroll
  for (int i = 0; i < 24; ++i) {
    int c = lane + 32 * i;
    float o = (v[i] - mu) * inv * gamma[c] + beta[c];
    if (outF) outF[(size_t)row * H_ + c] = o;
    if (outH) outH[(size_t)row * H_ + c] = (_Float16)o;
  }
}

// ---------- deg_inv = 1/max(rowsum(adj),1), wave-per-row ----------
__global__ __launch_bounds__(256) void deg_kernel(
    const float* __restrict__ adj, float* __restrict__ deginv, int nrows) {
  int wave = threadIdx.x >> 5, lane = threadIdx.x & 31;
  int row = blockIdx.x * 8 + wave;
  if (row >= nrows) return;
  const float* p = adj + (size_t)row * NN;
  float s = 0.f;
#pragma unroll
  for (int i = 0; i < 16; ++i) s += p[lane + 32 * i];
  for (int off = 16; off > 0; off >>= 1) s += __shfl_xor(s, off);
  if (lane == 0) deginv[row] = 1.f / fmaxf(s, 1.f);
}

// ---------- pooled[b,h] = mean_n x[b,n,h] ----------
__global__ __launch_bounds__(256) void pool_kernel(
    const float* __restrict__ x, float* __restrict__ pooled) {
  int b = blockIdx.y;
  int h = blockIdx.x * 256 + threadIdx.x;
  const float* p = x + (size_t)b * NN * H_ + h;
  float s = 0.f;
  for (int n = 0; n < NN; ++n) s += p[(size_t)n * H_];
  pooled[b * H_ + h] = s * (1.f / NN);
}

// ---------- gate: softmax(pooled @ gW + gb); also accumulate relation stat ----------
__global__ void gate_kernel(const float* __restrict__ pooled,
                            const float* __restrict__ gW, const float* __restrict__ gb,
                            float* __restrict__ w, float* __restrict__ stat, int initStat) {
  int b = threadIdx.x;  // 32 threads = B
  if (initStat && b < R_) stat[b] = 0.f;
  __syncthreads();
  const float* p = pooled + b * H_;
  float lg[R_];
#pragma unroll
  for (int r = 0; r < R_; ++r) lg[r] = gb[r];
  for (int h = 0; h < H_; ++h) {
    float pv = p[h];
#pragma unroll
    for (int r = 0; r < R_; ++r) lg[r] += pv * gW[h * R_ + r];
  }
  float m = lg[0];
#pragma unroll
  for (int r = 1; r < R_; ++r) m = fmaxf(m, lg[r]);
  float e[R_], s = 0.f;
#pragma unroll
  for (int r = 0; r < R_; ++r) { e[r] = __expf(lg[r] - m); s += e[r]; }
  float is = 1.f / s;
#pragma unroll
  for (int r = 0; r < R_; ++r) {
    float wr = e[r] * is;
    w[b * R_ + r] = wr;
    atomicAdd(&stat[r], wr * (1.f / (B_ * L_)));
  }
}

// ---------- f32 -> f16 conversion ----------
__global__ void cvt_kernel(const float* __restrict__ in, _Float16* __restrict__ out, long n) {
  long i = (long)blockIdx.x * blockDim.x + threadIdx.x;
  long stride = (long)gridDim.x * blockDim.x;
  for (; i < n; i += stride) out[i] = (_Float16)in[i];
}

extern "C" void kernel_launch(void* const* d_in, const int* in_sizes, int n_in,
                              void* d_out, int out_size, void* d_ws, size_t ws_size,
                              hipStream_t stream) {
  (void)in_sizes; (void)n_in; (void)out_size;
  const float* x0   = (const float*)d_in[0];
  const float* adj  = (const float*)d_in[1];
  const float* relW = (const float*)d_in[2];
  const float* relb = (const float*)d_in[3];
  const float* gW   = (const float*)d_in[4];
  const float* gb   = (const float*)d_in[5];
  const float* outW = (const float*)d_in[6];
  const float* outb = (const float*)d_in[7];
  const float* ln1g = (const float*)d_in[8];
  const float* ln1b = (const float*)d_in[9];
  const float* W1   = (const float*)d_in[10];
  const float* b1   = (const float*)d_in[11];
  const float* W2   = (const float*)d_in[12];
  const float* b2   = (const float*)d_in[13];
  const float* ln2g = (const float*)d_in[14];
  const float* ln2b = (const float*)d_in[15];

  float* out  = (float*)d_out;
  float* stat = out + (size_t)B_ * NN * H_;

  char* ws = (char*)d_ws;
  size_t off = 0;
  auto alloc = [&](size_t bytes) -> void* {
    void* p = ws + off;
    off += (bytes + 255) & ~(size_t)255;
    return p;
  };
  float*    deginv  = (float*)   alloc((size_t)B_ * R_ * NN * 4);
  _Float16* xh      = (_Float16*)alloc((size_t)B_ * NN * H_ * 2);   // also merged_h
  char*     aggB    = (char*)    alloc((size_t)B_ * R_ * NN * H_ * 2);
  _Float16* aggh    = (_Float16*)aggB;
  _Float16* hiddenh = (_Float16*)aggB;                              // reuse (agg dead after G2)
  _Float16* ff1h    = (_Float16*)(aggB + (size_t)B_ * NN * H_ * 2);
  float*    hidpre  = (float*)   alloc((size_t)B_ * NN * H_ * 4);
  float*    xbuf    = (float*)   alloc((size_t)B_ * NN * H_ * 4);
  float*    pooled  = (float*)   alloc((size_t)B_ * H_ * 4);
  float*    wgate   = (float*)   alloc((size_t)B_ * R_ * 4);
  _Float16* relWh   = (_Float16*)alloc((size_t)R_ * H_ * H_ * 2);
  _Float16* outWh   = (_Float16*)alloc((size_t)H_ * H_ * 2);
  _Float16* W1h     = (_Float16*)alloc((size_t)H_ * FF_ * 2);
  _Float16* W2h     = (_Float16*)alloc((size_t)FF_ * H_ * 2);
  if (off > ws_size) return;  // workspace too small; avoid corruption

  deg_kernel<<<(B_ * R_ * NN) / 8, 256, 0, stream>>>(adj, deginv, B_ * R_ * NN);

  const float* xcur = x0;
  for (int l = 0; l < L_; ++l) {
    cvt_kernel<<<2048, 256, 0, stream>>>(relW + (size_t)l * R_ * H_ * H_, relWh, (long)R_ * H_ * H_);
    cvt_kernel<<<1024, 256, 0, stream>>>(outW + (size_t)l * H_ * H_, outWh, (long)H_ * H_);
    cvt_kernel<<<1024, 256, 0, stream>>>(W1   + (size_t)l * H_ * FF_, W1h, (long)H_ * FF_);
    cvt_kernel<<<1024, 256, 0, stream>>>(W2   + (size_t)l * FF_ * H_, W2h, (long)FF_ * H_);
    cvt_kernel<<<4096, 256, 0, stream>>>(xcur, xh, (long)B_ * NN * H_);

    pool_kernel<<<dim3(H_ / 256, B_), 256, 0, stream>>>(xcur, pooled);
    gate_kernel<<<1, 32, 0, stream>>>(pooled, gW + (size_t)l * H_ * R_,
                                      gb + (size_t)l * R_, wgate, stat, l == 0);

    gemm_agg_kernel<<<dim3(H_ / BLK_N, NN / BLK_M, B_ * R_), NT, 0, stream>>>(
        adj, xh, deginv, aggh);

    gemm_merge_kernel<<<dim3(H_ / BLK_N, NN / BLK_M, B_), NT, 0, stream>>>(
        aggh, relWh, relb + (size_t)l * R_ * H_, wgate, xh /*merged_h*/);

    gemm_bias_kernel<<<dim3(H_ / BLK_N, NN / BLK_M, B_), NT, 0, stream>>>(
        xh, outWh, outb + (size_t)l * H_, xcur, hidpre, (_Float16*)nullptr,
        NN, H_, H_, (long)NN * H_, (long)NN * H_, (long)NN * H_, 0);

    ln_kernel<<<(B_ * NN) / 8, 256, 0, stream>>>(
        hidpre, ln1g + (size_t)l * H_, ln1b + (size_t)l * H_, hidpre, hiddenh, B_ * NN);

    gemm_bias_kernel<<<dim3(FF_ / BLK_N, NN / BLK_M, B_), NT, 0, stream>>>(
        hiddenh, W1h, b1 + (size_t)l * FF_, (const float*)nullptr,
        (float*)nullptr, ff1h,
        NN, FF_, H_, (long)NN * H_, 0L, (long)NN * FF_, 1);

    gemm_bias_kernel<<<dim3(H_ / BLK_N, NN / BLK_M, B_), NT, 0, stream>>>(
        ff1h, W2h, b2 + (size_t)l * H_, hidpre, hidpre, (_Float16*)nullptr,
        NN, H_, FF_, (long)NN * FF_, (long)NN * H_, (long)NN * H_, 0);

    ln_kernel<<<(B_ * NN) / 8, 256, 0, stream>>>(
        hidpre, ln2g + (size_t)l * H_, ln2b + (size_t)l * H_,
        (l == L_ - 1) ? out : xbuf, (_Float16*)nullptr, B_ * NN);

    xcur = xbuf;
  }
}
